// cFSMN_layer_72662256713826
// MI455X (gfx1250) — matrix operand order repaired and checked by reading, not compiled, not run on hardware
//
#include <hip/hip_runtime.h>
#include <hip/hip_bf16.h>

#define N_TOK 16384
#define DD    512
#define HH    2048
#define EE    8
#define EMBD  256

typedef __bf16 bfx16 __attribute__((ext_vector_type(16)));
typedef float  f32x8 __attribute__((ext_vector_type(8)));

// ---------------------------------------------------------------- init
__global__ void k_init(int* counts) {
    if (threadIdx.x < EE) counts[threadIdx.x] = 0;
}

// ---------------------------------------------------------------- weight pre-pack
// Repack f32 weights [E][kTiles*32][nTiles*16] into WMMA-B-fragment-native bf16:
// frag = (e*kTiles + kt)*nTiles + nt ; per frag, lane L holds 16 contiguous bf16:
//   n = nt*16 + (L&15), k = kt*32 + ((L>>4)<<4) + e16
__global__ void __launch_bounds__(256) k_pack(const float* __restrict__ W,
                                              __bf16* __restrict__ Wp,
                                              int kTiles, int nTiles) {
    const int gid  = blockIdx.x * 256 + threadIdx.x;
    const int lane = gid & 31;
    const int frag = gid >> 5;
    const int nt = frag % nTiles;
    const int r  = frag / nTiles;
    const int kt = r % kTiles;
    const int e  = r / kTiles;
    const int ld = nTiles * 16;
    const int k0 = kt * 32 + ((lane >> 4) << 4);
    const int n  = nt * 16 + (lane & 15);
    const float* src = W + ((size_t)(e * kTiles * 32 + k0)) * ld + n;
    union { uint4 u[2]; bfx16 v; } t;
#pragma unroll
    for (int q = 0; q < 16; ++q) { t.v[q] = (__bf16)(*src); src += ld; }
    uint4* dst = (uint4*)(Wp + ((size_t)frag * 32 + lane) * 16);
    dst[0] = t.u[0]; dst[1] = t.u[1];
}

// ---------------------------------------------------------------- router (top-1)
__global__ void __launch_bounds__(256) k_router(
    const float* __restrict__ x, const float* __restrict__ emb,
    const float* __restrict__ rw, float* __restrict__ gate_val,
    int* __restrict__ gate_idx, int* __restrict__ counts, int* __restrict__ lists) {
    __shared__ float wsm[(EMBD + DD) * EE];
    const int tid = threadIdx.x;
    for (int i = tid; i < (EMBD + DD) * EE; i += 256) wsm[i] = rw[i];
    __syncthreads();

    const int tok = blockIdx.x * 256 + tid;
    float acc[EE];
#pragma unroll
    for (int j = 0; j < EE; ++j) acc[j] = 0.f;
    const float* ep = emb + (size_t)tok * EMBD;
    for (int f = 0; f < EMBD; ++f) {
        float v = ep[f];
#pragma unroll
        for (int j = 0; j < EE; ++j) acc[j] += v * wsm[f * EE + j];
    }
    const float* xp = x + (size_t)tok * DD;
    for (int f = 0; f < DD; ++f) {
        float v = xp[f];
#pragma unroll
        for (int j = 0; j < EE; ++j) acc[j] += v * wsm[(EMBD + f) * EE + j];
    }
    float best = acc[0]; int bi = 0;
#pragma unroll
    for (int j = 1; j < EE; ++j) if (acc[j] > best) { best = acc[j]; bi = j; }
    float s = 0.f;
#pragma unroll
    for (int j = 0; j < EE; ++j) s += __expf(acc[j] - best);
    gate_val[tok] = 1.f / s;          // softmax max-prob = 1/sum(exp(l-lmax))
    gate_idx[tok] = bi;
    int slot = atomicAdd(&counts[bi], 1);
    lists[bi * N_TOK + slot] = tok;
}

// ---------------------------------------------------------------- WMMA fragment loaders
// A (16x32 bf16) from LDS, row-major [rows][ld] bf16 -> two ds_load_b128.
__device__ __forceinline__ bfx16 load_a_lds(const __bf16* base, int mOff, int kOff, int ld) {
    const int lane = threadIdx.x & 31;
    const __bf16* p = base + (size_t)(mOff + (lane & 15)) * ld + kOff + ((lane >> 4) << 4);
    union { uint4 u[2]; bfx16 v; } t;
    t.u[0] = *(const uint4*)(p);
    t.u[1] = *(const uint4*)(p + 8);
    return t.v;
}

// B fragment (32B per lane) from a pre-packed pointer -> two global_load_b128.
__device__ __forceinline__ bfx16 load_frag(const uint4* __restrict__ p) {
    union { uint4 u[2]; bfx16 v; } t;
    t.u[0] = p[0]; t.u[1] = p[1];
    return t.v;
}

// ---------------------------------------------------------------- grouped expert FFN
// grid (256, 8): 64-token groups per expert.  256 thr = 8 waves.
// Wave tiling: 2 M-tiles x 8 N-tiles; B fragment feeds two wmmas.
__global__ void __launch_bounds__(256) k_ffn(
    const float* __restrict__ x, const __bf16* __restrict__ W1p,
    const float* __restrict__ b1, const __bf16* __restrict__ W2p,
    const int* __restrict__ seq_len, const float* __restrict__ gate_val,
    const int* __restrict__ counts, const int* __restrict__ lists,
    float* __restrict__ p_buf) {
    extern __shared__ char smem[];
    __bf16* xT  = (__bf16*)smem;                       // 64x512 bf16
    __bf16* hT  = (__bf16*)(smem + 64 * DD * 2);       // 64x512 bf16
    int*   tokS = (int*)(smem + 2 * 64 * DD * 2);
    float* gS   = (float*)(tokS + 64);
    float* mS   = (float*)(gS + 64);

    const int ex   = blockIdx.y;
    const int cnt  = counts[ex];
    const int base = blockIdx.x * 64;
    if (base >= cnt) return;
    const int rows = min(64, cnt - base);
    const int tid  = threadIdx.x;

    if (tid < 64) {
        int tok = 0; float g = 0.f, msk = 0.f;
        if (tid < rows) {
            tok = lists[ex * N_TOK + base + tid];
            g   = gate_val[tok];
            int b = tok >> 10, t = tok & 1023;
            msk = (t < seq_len[b]) ? 1.f : 0.f;
        }
        tokS[tid] = tok; gS[tid] = g; mS[tid] = msk;
    }
    __syncthreads();

    // stage gathered x rows as bf16 (zero-pad invalid rows)
    for (int i = tid; i < 64 * (DD / 4); i += 256) {
        int m = i >> 7, c4 = i & 127;
        float4 v = make_float4(0.f, 0.f, 0.f, 0.f);
        if (m < rows) v = *(const float4*)(x + (size_t)tokS[m] * DD + c4 * 4);
        __bf16* dst = xT + (size_t)m * DD + c4 * 4;
        dst[0] = (__bf16)v.x; dst[1] = (__bf16)v.y;
        dst[2] = (__bf16)v.z; dst[3] = (__bf16)v.w;
    }
    __syncthreads();

    const int w     = tid >> 5;
    const int lane  = tid & 31;
    const int mA    = (w & 1) * 32;      // wave's two 16-row M tiles: mA, mA+16
    const int nBase = (w >> 1) * 128;    // wave's 128-wide N quarter (8 n-tiles)
    const int ntB   = nBase >> 4;        // first n-tile index
    const float* b1e = b1 + (size_t)ex * HH;

    f32x8 acc2[8][2];
#pragma unroll
    for (int j = 0; j < 8; ++j)
#pragma unroll
        for (int h = 0; h < 2; ++h)
#pragma unroll
            for (int v = 0; v < 8; ++v) acc2[j][h][v] = 0.f;

    for (int c = 0; c < 4; ++c) {        // H chunks of 512
        // GEMM1: hid_chunk[64,512] = relu(xT @ W1[:, c*512+...] + b1)
#pragma unroll
        for (int j = 0; j < 8; ++j) {
            const int nOff = nBase + j * 16;
            const uint4* bp = (const uint4*)(W1p +
                (((size_t)(ex * 2048 + c * 32 + ntB + j)) * 32 + lane) * 16);
            f32x8 a0, a1;
#pragma unroll
            for (int v = 0; v < 8; ++v) { a0[v] = 0.f; a1[v] = 0.f; }
            bfx16 af0 = load_a_lds(xT, mA,      0, DD);
            bfx16 af1 = load_a_lds(xT, mA + 16, 0, DD);
            bfx16 bf  = load_frag(bp);
#pragma unroll 3
            for (int k = 0; k < 15; ++k) {       // software pipeline: prefetch k+1
                bp += 8192;                      // +128 frags (1 KB each) / 16 B
                bfx16 nf0 = load_a_lds(xT, mA,      (k + 1) * 32, DD);
                bfx16 nf1 = load_a_lds(xT, mA + 16, (k + 1) * 32, DD);
                bfx16 nb  = load_frag(bp);
                a0 = __builtin_amdgcn_wmma_f32_16x16x32_bf16(
                         false, af0, false, bf, (short)0, a0, false, false);
                a1 = __builtin_amdgcn_wmma_f32_16x16x32_bf16(
                         false, af1, false, bf, (short)0, a1, false, false);
                af0 = nf0; af1 = nf1; bf = nb;
            }
            a0 = __builtin_amdgcn_wmma_f32_16x16x32_bf16(
                     false, af0, false, bf, (short)0, a0, false, false);
            a1 = __builtin_amdgcn_wmma_f32_16x16x32_bf16(
                     false, af1, false, bf, (short)0, a1, false, false);

            const float bias = b1e[c * 512 + nOff + (lane & 15)];
            const int mr = ((lane >> 4) << 3);
#pragma unroll
            for (int v = 0; v < 8; ++v) {
                float t0 = a0[v] + bias; t0 = t0 > 0.f ? t0 : 0.f;
                float t1 = a1[v] + bias; t1 = t1 > 0.f ? t1 : 0.f;
                hT[(size_t)(mA + mr + v) * DD + nOff + (lane & 15)]      = (__bf16)t0;
                hT[(size_t)(mA + 16 + mr + v) * DD + nOff + (lane & 15)] = (__bf16)t1;
            }
        }
        __syncthreads();
        // GEMM2 partial: acc2 += hid_chunk @ W2[c*512+..., :]
#pragma unroll
        for (int j = 0; j < 8; ++j) {
            const uint4* bp = (const uint4*)(W2p +
                (((size_t)(ex * 2048 + c * 512 + ntB + j)) * 32 + lane) * 16);
            bfx16 af0 = load_a_lds(hT, mA,      0, DD);
            bfx16 af1 = load_a_lds(hT, mA + 16, 0, DD);
            bfx16 bf  = load_frag(bp);
#pragma unroll 3
            for (int k = 0; k < 15; ++k) {       // software pipeline: prefetch k+1
                bp += 2048;                      // +32 frags (1 KB each) / 16 B
                bfx16 nf0 = load_a_lds(hT, mA,      (k + 1) * 32, DD);
                bfx16 nf1 = load_a_lds(hT, mA + 16, (k + 1) * 32, DD);
                bfx16 nb  = load_frag(bp);
                acc2[j][0] = __builtin_amdgcn_wmma_f32_16x16x32_bf16(
                                 false, af0, false, bf, (short)0, acc2[j][0], false, false);
                acc2[j][1] = __builtin_amdgcn_wmma_f32_16x16x32_bf16(
                                 false, af1, false, bf, (short)0, acc2[j][1], false, false);
                af0 = nf0; af1 = nf1; bf = nb;
            }
            acc2[j][0] = __builtin_amdgcn_wmma_f32_16x16x32_bf16(
                             false, af0, false, bf, (short)0, acc2[j][0], false, false);
            acc2[j][1] = __builtin_amdgcn_wmma_f32_16x16x32_bf16(
                             false, af1, false, bf, (short)0, acc2[j][1], false, false);
        }
        __syncthreads();
    }

    // scatter: p = (ffn*gate + x) * seq_mask
#pragma unroll
    for (int j = 0; j < 8; ++j) {
        const int nCol = nBase + j * 16 + (lane & 15);
#pragma unroll
        for (int h = 0; h < 2; ++h) {
            const int mrow = mA + h * 16 + ((lane >> 4) << 3);
#pragma unroll
            for (int v = 0; v < 8; ++v) {
                const int m = mrow + v;
                if (m < rows) {
                    const int tok = tokS[m];
                    float val = (acc2[j][h][v] * gS[m] + x[(size_t)tok * DD + nCol]) * mS[m];
                    p_buf[(size_t)tok * DD + nCol] = val;
                }
            }
        }
    }
}

// ---------------------------------------------------------------- 11-tap depthwise FIR + skip
__global__ void __launch_bounds__(256) k_fir(
    const float* __restrict__ p, const float* __restrict__ lf,
    const float* __restrict__ cf, const float* __restrict__ rf,
    float* __restrict__ out) {
    const int g  = blockIdx.x * 256 + threadIdx.x;   // over N_TOK * (D/4)
    const int d4 = g & 127;
    const int n  = g >> 7;
    const int b  = n >> 10, t = n & 1023;
    const float4* P = (const float4*)p;
    float4 acc = P[(size_t)n * 128 + d4];            // the "+ p" term
#pragma unroll
    for (int k = 0; k < 11; ++k) {
        const int tp = t + k - 5;
        if (tp >= 0 && tp < 1024) {
            float4 f;
            if (k < 5)       f = ((const float4*)lf)[k * 128 + d4];
            else if (k == 5) f = ((const float4*)cf)[d4];
            else             f = ((const float4*)rf)[(k - 6) * 128 + d4];
            float4 pv = P[(size_t)(b * 1024 + tp) * 128 + d4];
            acc.x += f.x * pv.x; acc.y += f.y * pv.y;
            acc.z += f.z * pv.z; acc.w += f.w * pv.w;
        }
    }
    ((float4*)out)[(size_t)n * 128 + d4] = acc;
}

// ---------------------------------------------------------------- launch
extern "C" void kernel_launch(void* const* d_in, const int* in_sizes, int n_in,
                              void* d_out, int out_size, void* d_ws, size_t ws_size,
                              hipStream_t stream) {
    const float* x   = (const float*)d_in[0];
    const float* emb = (const float*)d_in[1];
    const int*   seq = (const int*)d_in[2];
    const float* W1  = (const float*)d_in[3];
    const float* b1  = (const float*)d_in[4];
    const float* W2  = (const float*)d_in[5];
    const float* lf  = (const float*)d_in[6];
    const float* cf  = (const float*)d_in[7];
    const float* rf  = (const float*)d_in[8];
    const float* rw  = (const float*)d_in[9];
    float* out = (float*)d_out;

    char* ws = (char*)d_ws;
    int*    counts = (int*)ws;                                    // @0
    float*  gate_v = (float*)(ws + 256);                          // N f32
    int*    gate_i = (int*)(ws + 65792);                          // N i32
    int*    lists  = (int*)(ws + 131328);                         // E*N i32
    float*  p_buf  = (float*)(ws + 655616);                       // N*D f32
    __bf16* Wp1    = (__bf16*)(ws + 34210048);                    // E*D*H bf16
    __bf16* Wp2    = (__bf16*)(ws + 50987264);                    // E*H*D bf16

    k_init<<<1, 32, 0, stream>>>(counts);
    // W1: [E][512][2048] -> kTiles=16, nTiles=128 ; W2: [E][2048][512] -> 64, 32
    k_pack<<<EE * 16 * 128 / 8, 256, 0, stream>>>(W1, Wp1, 16, 128);
    k_pack<<<EE * 64 * 32 / 8, 256, 0, stream>>>(W2, Wp2, 64, 32);
    k_router<<<N_TOK / 256, 256, 0, stream>>>(x, emb, rw, gate_v, gate_i, counts, lists);

    const size_t shmem = 2 * 64 * DD * 2 + 64 * 12;   // xT + hT + tok/gate/mask
    dim3 gffn(256, EE);
    k_ffn<<<gffn, 256, shmem, stream>>>(x, Wp1, b1, Wp2, seq, gate_v, counts, lists, p_buf);

    k_fir<<<(N_TOK * (DD / 4)) / 256, 256, 0, stream>>>(p_buf, lf, cf, rf, out);
}
